// V8Model_13752485282513
// MI455X (gfx1250) — matrix operand order repaired and checked by reading, not compiled, not run on hardware
//
#include <hip/hip_runtime.h>
#include <hip/hip_bf16.h>

// ---- problem constants (match reference) ----
#define BSZ   2
#define TT    2048
#define VOCAB 32000
#define DM    1024
#define CCH   8          // C
#define MPB   64         // M
#define NNEUR 512        // N = C*M
#define KNB   16         // K neighbors
#define DECAYF 0.9f

typedef __attribute__((ext_vector_type(16))) __bf16 v16bf;
typedef __attribute__((ext_vector_type(8)))  float  v8f;

static __device__ __forceinline__ unsigned f32_to_bf16_bits(float f) {
  union { float f; unsigned u; } v; v.f = f;
  unsigned r = v.u + 0x7FFFu + ((v.u >> 16) & 1u);   // round-to-nearest-even
  return r >> 16;
}
static __device__ __forceinline__ unsigned pack2bf(float lo, float hi) {
  return f32_to_bf16_bits(lo) | (f32_to_bf16_bits(hi) << 16);
}

// ---------------------------------------------------------------------------
// Kernel: embedding gather  H[row,:] = E[ids[row],:]
// ---------------------------------------------------------------------------
__global__ void embed_gather(const int* __restrict__ ids,
                             const float* __restrict__ E,
                             float* __restrict__ H) {
  const int row = blockIdx.x;
  const float4* src = (const float4*)(E + (size_t)ids[row] * DM);
  float4* dst = (float4*)(H + (size_t)row * DM);
  for (int i = threadIdx.x; i < DM / 4; i += blockDim.x) dst[i] = src[i];
}

// ---------------------------------------------------------------------------
// Generic f32-in/f32-out GEMM via v_wmma_f32_16x16x32_bf16.
// out[M,N] = epi( A[M,K] @ B[K,N] (+Cadd) ), epi = tanh if doTanh.
// Workgroup = 128 threads (4 waves). Tile: 128(M) x 64(N), K-step 32.
// Wave w computes rows [w*32, w*32+32) as 2 strips x 4 col-tiles = 8 WMMA.
// LDS layout is fragment-friendly:
//   sA [row][k]   row-stride 40 ushorts (80B = 5*16B -> aligned b128 chunks)
//   sBt[col][k]   B stored transposed so a lane's 16 K-values are contiguous
// Staging is vectorized (b128 global loads / b64 LDS stores) and register-
// prefetched so global latency overlaps the WMMA work on the current tile.
// ---------------------------------------------------------------------------
#define LDK 40   // padded K-stride (ushorts)

__global__ __launch_bounds__(128)
void wmma_gemm_bf16(const float* __restrict__ A, const float* __restrict__ B,
                    const float* __restrict__ Cadd, float* __restrict__ out,
                    int K, int N, int doTanh) {
  __shared__ unsigned short sA[128][LDK];   // 128x32 bf16 A tile
  __shared__ unsigned short sBt[64][LDK];   // 32x64  bf16 B tile (transposed)

  const int tid  = threadIdx.x;
  const int lane = tid & 31;
  const int wave = tid >> 5;          // 0..3
  const int grp  = lane >> 4;         // lane half -> K-range selector
  const int l16  = lane & 15;
  const int tileN = blockIdx.x * 64;
  const int tileM = blockIdx.y * 128;

  v8f zero;
  #pragma unroll
  for (int i = 0; i < 8; i++) zero[i] = 0.0f;
  v8f acc[2][4] = {{zero, zero, zero, zero}, {zero, zero, zero, zero}};

  float4 ra[8];      // A: 128x32 / 128 thr = 8 float4 per thread
  float  rb[4][4];   // B: 8(kq) x 64(n) quads / 128 thr = 4 quads (4 k each)

  auto load_regs = [&](int k0) {
    #pragma unroll
    for (int j = 0; j < 8; j++) {                    // A, b128 loads
      const int i4 = tid + j * 128;                  // quad id in [0,1024)
      const int r = i4 >> 3, c = (i4 & 7) * 4;
      ra[j] = *(const float4*)&A[(size_t)(tileM + r) * K + k0 + c];
    }
    #pragma unroll
    for (int j = 0; j < 4; j++) {                    // B, lane-coalesced b32
      const int i4 = tid + j * 128;                  // quad id in [0,512)
      const int n = i4 & 63, kq = i4 >> 6;           // quad = 4 K's, fixed n
      #pragma unroll
      for (int q = 0; q < 4; q++)
        rb[j][q] = B[(size_t)(k0 + kq * 4 + q) * N + tileN + n];
    }
  };
  auto store_lds = [&]() {
    #pragma unroll
    for (int j = 0; j < 8; j++) {                    // A -> ds_store_b64
      const int i4 = tid + j * 128;
      const int r = i4 >> 3, c = (i4 & 7) * 4;
      uint2 p; p.x = pack2bf(ra[j].x, ra[j].y); p.y = pack2bf(ra[j].z, ra[j].w);
      *(uint2*)&sA[r][c] = p;
    }
    #pragma unroll
    for (int j = 0; j < 4; j++) {                    // B -> ds_store_b64 (transposed)
      const int i4 = tid + j * 128;
      const int n = i4 & 63, kq = i4 >> 6;
      uint2 p; p.x = pack2bf(rb[j][0], rb[j][1]); p.y = pack2bf(rb[j][2], rb[j][3]);
      *(uint2*)&sBt[n][kq * 4] = p;
    }
  };

  union Frag { v16bf bf; uint4 q[2]; };

  load_regs(0);
  store_lds();
  __syncthreads();

  const int nk = K >> 5;
  for (int kc = 0; kc < nk; kc++) {
    if (kc + 1 < nk) load_regs((kc + 1) << 5);  // in flight during WMMA work

    // A fragments: rows wave*32 + s*16 + l16; K chunks {8g..8g+7, 16+8g..+7}
    Frag af[2];
    #pragma unroll
    for (int s = 0; s < 2; s++) {
      const int am = wave * 32 + s * 16 + l16;
      af[s].q[0] = *(const uint4*)&sA[am][8 * grp];
      af[s].q[1] = *(const uint4*)&sA[am][16 + 8 * grp];
    }
    #pragma unroll
    for (int nt = 0; nt < 4; nt++) {
      // B fragment: col nt*16+l16, K chunks {8g..8g+7, 16+8g..+7}
      Frag bf;
      const int bn = nt * 16 + l16;
      bf.q[0] = *(const uint4*)&sBt[bn][8 * grp];
      bf.q[1] = *(const uint4*)&sBt[bn][16 + 8 * grp];
      #pragma unroll
      for (int s = 0; s < 2; s++) {
        acc[s][nt] = __builtin_amdgcn_wmma_f32_16x16x32_bf16(
            false, af[s].bf, false, bf.bf, (short)0, acc[s][nt], false, false);
      }
    }
    __syncthreads();
    if (kc + 1 < nk) store_lds();
    __syncthreads();
  }

  // epilogue: D layout — VGPR r, lanes 0-15 -> M=r, lanes 16-31 -> M=8+r
  #pragma unroll
  for (int s = 0; s < 2; s++) {
    const int rowBase = tileM + wave * 32 + s * 16 + 8 * grp;
    #pragma unroll
    for (int nt = 0; nt < 4; nt++) {
      #pragma unroll
      for (int r = 0; r < 8; r++) {
        const size_t idx = (size_t)(rowBase + r) * N + tileN + nt * 16 + l16;
        float v = acc[s][nt][r];
        if (Cadd) v += Cadd[idx];
        if (doTanh) v = tanhf(v);
        out[idx] = v;
      }
    }
  }
}

// ---------------------------------------------------------------------------
// aux loss = mean(surprise^2)
// ---------------------------------------------------------------------------
__global__ void aux_init(float* aux) { *aux = 0.0f; }

__global__ __launch_bounds__(256)
void aux_reduce(const float* __restrict__ surp, float* __restrict__ aux, int n) {
  __shared__ float sred[256];
  const int base = blockIdx.x * blockDim.x * 4 + threadIdx.x;
  float s = 0.0f;
  #pragma unroll
  for (int j = 0; j < 4; j++) {
    const int idx = base + j * 256;
    if (idx < n) { const float v = surp[idx]; s += v * v; }
  }
  sred[threadIdx.x] = s;
  __syncthreads();
  for (int off = 128; off > 0; off >>= 1) {
    if (threadIdx.x < off) sred[threadIdx.x] += sred[threadIdx.x + off];
    __syncthreads();
  }
  if (threadIdx.x == 0) atomicAdd(aux, sred[0] / (float)n);
}

// ---------------------------------------------------------------------------
// Recurrent scan: 1 persistent workgroup per batch element, mem state in LDS.
// Key identity: einsum('bnkd,d->bnk', mem[:,nbr], w_gate) == s[nbr] with
// s[m] = dot(mem[m], w_gate), so gates need only 512 row-dots per step.
// LDS mem rows padded to stride 65 so neighbor-row gathers spread banks.
// ---------------------------------------------------------------------------
#define MSTRIDE 65
__global__ __launch_bounds__(NNEUR)
void recurrent_scan(const float* __restrict__ cc,        // (BS*T, 512)
                    const float* __restrict__ W_in,      // 64x64
                    const float* __restrict__ w_gate,    // 64
                    const float* __restrict__ W_msg,     // 64x64
                    const float* __restrict__ thresh,    // BS x 512 x 16
                    const float* __restrict__ mem0,      // BS x 512 x 64
                    const int*   __restrict__ neighbors, // 512 x 16
                    float* __restrict__ mem_sig)         // (BS*T, 512)
{
  const int b   = blockIdx.x;
  const int tid = threadIdx.x;           // == neuron id n
  __shared__ float s_mem[NNEUR * MSTRIDE];  // 133 KB resident state
  __shared__ float s_Win[64 * 64];
  __shared__ float s_Wmsg[64 * 64];
  __shared__ float s_dot[NNEUR];
  __shared__ float s_inp[CCH * 64];
  __shared__ float s_cct[CCH * 64];
  __shared__ float s_g[64];

  for (int i = tid; i < 64 * 64; i += NNEUR) { s_Win[i] = W_in[i]; s_Wmsg[i] = W_msg[i]; }
  if (tid < 64) s_g[tid] = w_gate[tid];
  for (int d = 0; d < 64; d++)
    s_mem[tid * MSTRIDE + d] = mem0[((size_t)b * NNEUR + tid) * 64 + d];

  int   nb[KNB];
  float th[KNB];
  #pragma unroll
  for (int k = 0; k < KNB; k++) {
    nb[k] = neighbors[tid * KNB + k];
    th[k] = thresh[((size_t)b * NNEUR + tid) * KNB + k];
  }
  const int cI = tid >> 6;               // channel / block index (n // M)
  const int eI = tid & 63;
  __syncthreads();

  for (int t = 0; t < TT; t++) {
    // stage cc_t (8x64 = 512 values, one per thread)
    s_cct[tid] = cc[((size_t)(b * TT + t)) * NNEUR + tid];
    __syncthreads();

    // row-dots s[n] and inp[c][e] = cc_t[c,:] @ W_in[:,e]
    {
      float d0 = 0.0f;
      #pragma unroll
      for (int d = 0; d < 64; d++) d0 += s_mem[tid * MSTRIDE + d] * s_g[d];
      s_dot[tid] = d0;
      float ip = 0.0f;
      #pragma unroll
      for (int d = 0; d < 64; d++) ip += s_cct[cI * 64 + d] * s_Win[d * 64 + eI];
      s_inp[tid] = ip;                   // indexed as [c*64+e]
    }
    __syncthreads();

    // msg[n,:] = sum_k sigmoid(s[nbr]-thresh) * mem[nbr,:]
    float msg[64];
    #pragma unroll
    for (int d = 0; d < 64; d++) msg[d] = 0.0f;
    #pragma unroll
    for (int k = 0; k < KNB; k++) {
      const int r = nb[k];
      const float g = 1.0f / (1.0f + __expf(-(s_dot[r] - th[k])));
      const float* mrow = &s_mem[r * MSTRIDE];
      #pragma unroll
      for (int d = 0; d < 64; d++) msg[d] += g * mrow[d];
    }
    __syncthreads();  // all gathers done before state overwrite

    // new_mem[n,e] = tanh(0.9*mem + inp[blk(n)] + msg @ W_msg)
    float newrow[64];
    for (int e = 0; e < 64; e++) {
      float a = DECAYF * s_mem[tid * MSTRIDE + e] + s_inp[cI * 64 + e];
      #pragma unroll
      for (int d = 0; d < 64; d++) a += msg[d] * s_Wmsg[d * 64 + e];
      newrow[e] = tanhf(a);
    }
    #pragma unroll
    for (int e = 0; e < 64; e++) s_mem[tid * MSTRIDE + e] = newrow[e];
    __syncthreads();

    // mem_sig[b,t,c*64+e] = mean_m new_mem[c*64+m, e]
    float acc = 0.0f;
    #pragma unroll
    for (int m = 0; m < MPB; m++) acc += s_mem[(cI * 64 + m) * MSTRIDE + eI];
    mem_sig[((size_t)(b * TT + t)) * NNEUR + tid] = acc * (1.0f / 64.0f);
    __syncthreads();
  }
}

// ---------------------------------------------------------------------------
// launch
// ---------------------------------------------------------------------------
extern "C" void kernel_launch(void* const* d_in, const int* in_sizes, int n_in,
                              void* d_out, int out_size, void* d_ws, size_t ws_size,
                              hipStream_t stream) {
  const int*   ids     = (const int*)  d_in[0];
  const float* E       = (const float*)d_in[1];
  const float* W_surp  = (const float*)d_in[2];
  const float* W_cc    = (const float*)d_in[3];
  const float* W_in    = (const float*)d_in[4];
  const float* w_gate  = (const float*)d_in[5];
  const float* W_msg   = (const float*)d_in[6];
  const float* W_mem2h = (const float*)d_in[7];
  const float* W_out   = (const float*)d_in[8];
  const float* thresh  = (const float*)d_in[9];
  const float* mem0    = (const float*)d_in[10];
  const int*   nbrs    = (const int*)  d_in[11];

  const size_t ROWS = (size_t)BSZ * TT;          // 4096
  float* out    = (float*)d_out;
  float* logits = out;                            // ROWS*VOCAB f32
  float* aux    = out + ROWS * VOCAB;             // 1 f32
  float* surp   = aux + 1;                        // ROWS*512 f32

  float* H      = (float*)d_ws;                   // ROWS*1024
  float* cc     = H + ROWS * DM;                  // ROWS*512
  float* memsig = cc + ROWS * NNEUR;              // ROWS*512

  // 1) H = E[ids]
  embed_gather<<<dim3((unsigned)ROWS), 256, 0, stream>>>(ids, E, H);
  aux_init<<<1, 1, 0, stream>>>(aux);

  // 2) surprise = tanh(H @ W_surp)   -> d_out
  wmma_gemm_bf16<<<dim3(NNEUR / 64, ROWS / 128), 128, 0, stream>>>(
      H, W_surp, nullptr, surp, DM, NNEUR, 1);

  // 3) aux = mean(surprise^2)
  {
    const int n = (int)(ROWS * NNEUR);            // 2,097,152
    aux_reduce<<<(n + 1023) / 1024, 256, 0, stream>>>(surp, aux, n);
  }

  // 4) cc = H @ W_cc + surprise     -> ws
  wmma_gemm_bf16<<<dim3(NNEUR / 64, ROWS / 128), 128, 0, stream>>>(
      H, W_cc, surp, cc, DM, NNEUR, 0);

  // 5) sequential scan (persistent LDS state, one WG per batch)
  recurrent_scan<<<BSZ, NNEUR, 0, stream>>>(
      cc, W_in, w_gate, W_msg, thresh, mem0, nbrs, memsig);

  // 6) H += mem_sig @ W_mem2h  (in place; each WG touches only its own tile)
  wmma_gemm_bf16<<<dim3(DM / 64, ROWS / 128), 128, 0, stream>>>(
      memsig, W_mem2h, H, H, NNEUR, DM, 0);

  // 7) logits = H @ W_out  (dominant GEMM: 268 GFLOP)
  wmma_gemm_bf16<<<dim3(VOCAB / 64, ROWS / 128), 128, 0, stream>>>(
      H, W_out, nullptr, logits, DM, VOCAB, 0);
}
